// GATLayer_57921928954156
// MI455X (gfx1250) — matrix operand order repaired and checked by reading, not compiled, not run on hardware
//
#include <hip/hip_runtime.h>
#include <hip/hip_bf16.h>

#define NN      4096
#define IN_F    512
#define OUT_F   64
#define N_HEADS 8
#define ALPHA   0.2f
#define NEG_INF -9.0e15f

typedef __attribute__((ext_vector_type(2)))  float  v2f;
typedef __attribute__((ext_vector_type(8)))  float  v8f;
typedef __attribute__((ext_vector_type(16))) __bf16 v16bf;

union ABits { unsigned u[8]; v16bf v; };

// round-to-nearest-even float -> bf16 bits
__device__ __forceinline__ unsigned short f2bf(float x) {
  unsigned u = __builtin_bit_cast(unsigned, x);
  unsigned r = ((u >> 16) & 1u) + 0x7fffu;
  return (unsigned short)((u + r) >> 16);
}
__device__ __forceinline__ unsigned pack2bf(float lo, float hi) {
  return (unsigned)f2bf(lo) | ((unsigned)f2bf(hi) << 16);
}

// ---------------------------------------------------------------------------
// Kernel 1: Wh[4096,512] = h[4096,512] @ W[512,512]   (f32 WMMA 16x16x4)
// One wave per (16-row i-tile, head) -> 16x64 output tile, 4 N-subtiles.
// ---------------------------------------------------------------------------
__global__ __launch_bounds__(128) void k1_gemm_wh(const float* __restrict__ h,
                                                  const float* __restrict__ W,
                                                  float* __restrict__ Wh) {
  const int wave = blockIdx.x * (blockDim.x >> 5) + (threadIdx.x >> 5);
  const int lane = threadIdx.x & 31;
  const int head = wave & 7;
  const int i0   = (wave >> 3) << 4;
  const int mrow  = lane & 15;          // A: row M ; B/C/D: col N
  const int khalf = (lane >> 4) << 1;   // f32 K split across lane halves

  v8f acc[4] = {};
  const float* hrow = h + (size_t)(i0 + mrow) * IN_F;
  const int cb = head * OUT_F + mrow;

  for (int k0 = 0; k0 < IN_F; k0 += 4) {
    v2f a;                              // A 16x4: V0=K(khalf), V1=K(khalf+1)
    a.x = hrow[k0 + khalf];
    a.y = hrow[k0 + khalf + 1];
    const int kb = k0 + khalf;          // B 4x16: same lane-half K split
#pragma unroll
    for (int nt = 0; nt < 4; ++nt) {
      v2f b;
      b.x = W[(size_t)kb * IN_F + cb + nt * 16];
      b.y = W[(size_t)(kb + 1) * IN_F + cb + nt * 16];
      acc[nt] = __builtin_amdgcn_wmma_f32_16x16x4_f32(
          false, a, false, b, (short)0, acc[nt], false, false);
    }
  }
  const int rb = (lane >> 4) << 3;      // C/D: M = r + 8*(lane>=16)
#pragma unroll
  for (int nt = 0; nt < 4; ++nt)
#pragma unroll
    for (int r = 0; r < 8; ++r)
      Wh[(size_t)(i0 + rb + r) * IN_F + head * OUT_F + nt * 16 + mrow] = acc[nt][r];
}

// ---------------------------------------------------------------------------
// Kernel 2: per node n: stage Wh row, emit bf16 transposed WhT[f][n],
//           compute src[n,h] = Wh[n,h,:]·a_src ; dst[n,h] = Wh[n,h,:]·a_dst
// ---------------------------------------------------------------------------
__global__ __launch_bounds__(256) void k2_prep(const float* __restrict__ Wh,
                                               const float* __restrict__ a,
                                               unsigned short* __restrict__ WhT,
                                               float* __restrict__ srcv,
                                               float* __restrict__ dstv) {
  __shared__ float row[IN_F];
  const int n = blockIdx.x, t = threadIdx.x;
  for (int f = t; f < IN_F; f += 256) {
    float v = Wh[(size_t)n * IN_F + f];
    row[f] = v;
    WhT[(size_t)f * NN + n] = f2bf(v);
  }
  __syncthreads();
  if (t < 16) {
    const int hd = t & 7;
    const float* av = a + ((t >> 3) << 6);  // a_src (t<8) or a_dst (t>=8)
    float s = 0.f;
#pragma unroll 8
    for (int f = 0; f < OUT_F; ++f) s += row[hd * OUT_F + f] * av[f];
    (t < 8 ? srcv : dstv)[n * N_HEADS + hd] = s;
  }
}

// ---------------------------------------------------------------------------
// Kernel 3: masked online softmax stats per (row, head): rmax, 1/sum.
// One wave per row; 8 heads per lane; shfl_xor butterfly merge.
// ---------------------------------------------------------------------------
__global__ __launch_bounds__(256) void k3_stats(const int* __restrict__ adj,
                                                const float* __restrict__ srcv,
                                                const float* __restrict__ dstv,
                                                float* __restrict__ rmax,
                                                float* __restrict__ rinv) {
  const int i    = blockIdx.x * 8 + (threadIdx.x >> 5);
  const int lane = threadIdx.x & 31;
  float sm[8], m[8], s[8];
#pragma unroll
  for (int hd = 0; hd < 8; ++hd) {
    sm[hd] = srcv[i * N_HEADS + hd];
    m[hd]  = -__builtin_inff();
    s[hd]  = 0.f;
  }
  const int* arow = adj + (size_t)i * NN;
  for (int j = lane; j < NN; j += 32) {
    const int av = arow[j];
    const float4 d0 = ((const float4*)(dstv + j * N_HEADS))[0];
    const float4 d1 = ((const float4*)(dstv + j * N_HEADS))[1];
    const float dv[8] = {d0.x, d0.y, d0.z, d0.w, d1.x, d1.y, d1.z, d1.w};
#pragma unroll
    for (int hd = 0; hd < 8; ++hd) {
      float e = sm[hd] + dv[hd];
      e = e > 0.f ? e : ALPHA * e;
      e = av > 0 ? e : NEG_INF;
      const float mn = fmaxf(m[hd], e);
      s[hd] = s[hd] * __expf(m[hd] - mn) + __expf(e - mn);
      m[hd] = mn;
    }
  }
#pragma unroll
  for (int off = 16; off > 0; off >>= 1)
#pragma unroll
    for (int hd = 0; hd < 8; ++hd) {
      const float mo = __shfl_xor(m[hd], off, 32);
      const float so = __shfl_xor(s[hd], off, 32);
      const float mn = fmaxf(m[hd], mo);
      s[hd] = s[hd] * __expf(m[hd] - mn) + so * __expf(mo - mn);
      m[hd] = mn;
    }
  if (lane == 0)
#pragma unroll
    for (int hd = 0; hd < 8; ++hd) {
      rmax[i * N_HEADS + hd] = m[hd];
      rinv[i * N_HEADS + hd] = 1.f / s[hd];
    }
}

// ---------------------------------------------------------------------------
// Kernel 4: out = elu( softmax(e) @ Wh )  fused, bf16 WMMA 16x16x32.
// Block = 256 thr = 8 waves (one head each) x 32 rows; adj/dst tiles in LDS.
// ---------------------------------------------------------------------------
__global__ __launch_bounds__(256) void k4_aggr(const int* __restrict__ adj,
                                               const unsigned short* __restrict__ WhT,
                                               const float* __restrict__ srcv,
                                               const float* __restrict__ dstv,
                                               const float* __restrict__ rmax,
                                               const float* __restrict__ rinv,
                                               float* __restrict__ out) {
  __shared__ int   sadj[32 * 32];
  __shared__ float sdst[32 * N_HEADS];
  const int t    = threadIdx.x;
  const int hd   = t >> 5;              // wave id == head
  const int lane = t & 31;
  const int i0   = blockIdx.x * 32;
  const int mrow = lane & 15;           // A row / B col / D col
  const int klo  = (lane >> 4) << 3;    // bf16 A: lane-half K-pair offset (0/8); K offset 2*klo = 0/16

  float srcm[2], maxm[2], invm[2];
#pragma unroll
  for (int mt = 0; mt < 2; ++mt) {
    const int r = i0 + mt * 16 + mrow;
    srcm[mt] = srcv[r * N_HEADS + hd];
    maxm[mt] = rmax[r * N_HEADS + hd];
    invm[mt] = rinv[r * N_HEADS + hd];
  }

  v8f acc[2][4] = {};
  const unsigned* WT = (const unsigned*)WhT;   // bf16 pairs, ld = NN/2 dwords
  const int rr = t >> 3;                       // coop-load row 0..31
  const int cc = (t & 7) << 2;                 // coop-load col 0,4..28

  for (int j0 = 0; j0 < NN; j0 += 32) {
    // cooperative stage: 32x32 adj ints + 32x8 dst floats
    *(int4*)(sadj + rr * 32 + cc) =
        *(const int4*)(adj + (size_t)(i0 + rr) * NN + j0 + cc);
    sdst[t] = dstv[(j0 + (t >> 3)) * N_HEADS + (t & 7)];
    if (j0 + 32 < NN)
      __builtin_prefetch(adj + (size_t)(i0 + rr) * NN + j0 + 32 + cc, 0, 1);
    __syncthreads();

    // B fragments: 32x16 bf16, K pairs contiguous in WhT (dword loads)
    ABits bf[4];
#pragma unroll
    for (int nt = 0; nt < 4; ++nt) {
      const unsigned* bp =
          WT + (size_t)(hd * OUT_F + nt * 16 + mrow) * (NN / 2) + (j0 >> 1);
#pragma unroll
      for (int v = 0; v < 8; ++v)
        bf[nt].u[v] = bp[v + ((lane >> 4) << 3)];   // K = 2v + 16*(lane>=16)
    }

#pragma unroll
    for (int mt = 0; mt < 2; ++mt) {
      // A fragment: 16 probabilities, packed bf16, K = 2v + 16*(lane>=16) (matches B)
      ABits af;
      const int arow = mt * 16 + mrow;
#pragma unroll
      for (int v = 0; v < 8; ++v) {
        const int k0a = 2 * klo + 2 * v;
        float p[2];
#pragma unroll
        for (int q = 0; q < 2; ++q) {
          const int jl = k0a + q;
          float e = srcm[mt] + sdst[jl * N_HEADS + hd];
          e = e > 0.f ? e : ALPHA * e;
          e = sadj[arow * 32 + jl] > 0 ? e : NEG_INF;
          p[q] = __expf(e - maxm[mt]) * invm[mt];
        }
        af.u[v] = pack2bf(p[0], p[1]);
      }
#pragma unroll
      for (int nt = 0; nt < 4; ++nt)
        acc[mt][nt] = __builtin_amdgcn_wmma_f32_16x16x32_bf16(
            false, af.v, false, bf[nt].v, (short)0, acc[mt][nt], false, false);
    }
    __syncthreads();
  }

  // epilogue: ELU + store
  const int rb = (lane >> 4) << 3;
#pragma unroll
  for (int mt = 0; mt < 2; ++mt)
#pragma unroll
    for (int nt = 0; nt < 4; ++nt)
#pragma unroll
      for (int r = 0; r < 8; ++r) {
        const int row = i0 + mt * 16 + rb + r;
        float x = acc[mt][nt][r];
        x = x > 0.f ? x : (__expf(x) - 1.f);
        out[(size_t)row * (N_HEADS * OUT_F) + hd * OUT_F + nt * 16 + mrow] = x;
      }
}

// ---------------------------------------------------------------------------
extern "C" void kernel_launch(void* const* d_in, const int* in_sizes, int n_in,
                              void* d_out, int out_size, void* d_ws, size_t ws_size,
                              hipStream_t stream) {
  const float* h   = (const float*)d_in[0];
  const int*   adj = (const int*)d_in[1];
  const float* W   = (const float*)d_in[2];
  const float* a   = (const float*)d_in[3];
  float* out = (float*)d_out;

  char* ws = (char*)d_ws;
  float*          Wh   = (float*)ws;                                  // 8 MB
  unsigned short* WhT  = (unsigned short*)(ws + (size_t)(8u << 20));  // 4 MB
  float*          srcv = (float*)(ws + (size_t)(12u << 20));          // 128 KB
  float*          dstv = srcv + (size_t)NN * N_HEADS;
  float*          rmax = dstv + (size_t)NN * N_HEADS;
  float*          rinv = rmax + (size_t)NN * N_HEADS;

  k1_gemm_wh<<<512, 128, 0, stream>>>(h, W, Wh);
  k2_prep  <<<NN, 256, 0, stream>>>(Wh, a, WhT, srcv, dstv);
  k3_stats <<<NN / 8, 256, 0, stream>>>(adj, srcv, dstv, rmax, rinv);
  k4_aggr  <<<NN / 32, 256, 0, stream>>>(adj, WhT, srcv, dstv, rmax, rinv, out);
}
